// Encoder_41472204210373
// MI455X (gfx1250) — matrix operand order repaired and checked by reading, not compiled, not run on hardware
//
#include <hip/hip_runtime.h>
#include <hip/hip_bf16.h>
#include <stdint.h>

// ---------------------------------------------------------------------------
// CDNA5 (gfx1250) hierarchical-LSTM encoder.
//   - All GEMMs via v_wmma_f32_16x16x32_f16 (wave32 WMMA, f32 accumulate).
//   - Weights converted once per call to [N,K] f16 (B-fragment-contiguous).
//   - Per-step: GEMM1 ([v_t|h1] x Wcat1 -> gates1|zpre), cell1 (+gate),
//               GEMM2 ([h1*s|h2] x Wcat2 -> gates2), cell2.
//   - Epilogue behavior (bias / f32-out / f16-out) is template-selected so
//     no runtime null checks or per-element branches survive codegen.
// ---------------------------------------------------------------------------

typedef _Float16 half8  __attribute__((ext_vector_type(8)));
typedef _Float16 v16h   __attribute__((ext_vector_type(16)));
typedef float    v8f    __attribute__((ext_vector_type(8)));
typedef float    float8 __attribute__((ext_vector_type(8)));

#define FRAME_ 2048
#define PROJ_  512
#define MID_   128
#define HID_   1024
#define TT_    48
#define BB_    512

#define N1_ 4224   // 4*HID + MID   (gates1 | zpre)
#define K1_ 1536   // PROJ + HID    ([v_t | h1])
#define N2_ 4096   // 4*HID
#define K2_ 2048   // HID + HID     ([h1*s | h2])

// ---------------------------------------------------------------------------
// WMMA GEMM: C[M,N] = A[M,K] * B[N,K]^T (+bias[N]); B row-major [N,K] f16.
// Block = 256 threads (8 waves). Block tile 128x128. Wave tile 32x64.
// Grid = (N/128, M/128). All dims are multiples of the tile here.
// ---------------------------------------------------------------------------
template <bool AF32, bool HASB, bool OUTF, bool OUTH>
__global__ __launch_bounds__(256) void wmma_gemm(
    const void* __restrict__ Av, const _Float16* __restrict__ Bw,
    const float* __restrict__ bias,
    float* __restrict__ outF, _Float16* __restrict__ outH,
    int N, int K, int lda) {
  const int tid   = threadIdx.x;
  const int lane  = tid & 31;
  const int w     = tid >> 5;
  const int waveM = w & 3;       // 4 M-groups of 32 rows
  const int waveN = w >> 2;      // 2 N-groups of 64 cols
  const int rowBase = blockIdx.y * 128 + waveM * 32;
  const int colBase = blockIdx.x * 128 + waveN * 64;

  const int lrow  = lane & 15;   // row/col within a 16-wide tile
  const int lhalf = lane >> 4;   // K-half select per ISA fragment layout

  v8f acc[2][4];
#pragma unroll
  for (int mi = 0; mi < 2; ++mi)
#pragma unroll
    for (int ni = 0; ni < 4; ++ni) acc[mi][ni] = (v8f)0.0f;

  for (int k0 = 0; k0 < K; k0 += 32) {
    // A fragments: 16-bit A 16x32 layout — lanes 0-15: K {0..7,16..23},
    // lanes 16-31: K {8..15,24..31}.
    v16h afrag[2];
#pragma unroll
    for (int mi = 0; mi < 2; ++mi) {
      const int row = rowBase + mi * 16 + lrow;
      const int kb  = k0 + lhalf * 8;
      half8 l8, h8;
      if (AF32) {
        const float* Ap = (const float*)Av + (size_t)row * lda;
        float8 lo = *(const float8*)(Ap + kb);
        float8 hi = *(const float8*)(Ap + kb + 16);
        l8 = __builtin_convertvector(lo, half8);
        h8 = __builtin_convertvector(hi, half8);
      } else {
        const _Float16* Ap = (const _Float16*)Av + (size_t)row * lda;
        l8 = *(const half8*)(Ap + kb);
        h8 = *(const half8*)(Ap + kb + 16);
      }
      afrag[mi] = __builtin_shufflevector(l8, h8, 0, 1, 2, 3, 4, 5, 6, 7, 8, 9,
                                          10, 11, 12, 13, 14, 15);
    }
    // B fragments: lane l holds column n=l%16; lanes 0-15 K {0..7,16..23},
    // lanes 16-31 K {8..15,24..31} — same K order as A.
#pragma unroll
    for (int ni = 0; ni < 4; ++ni) {
      const int n = colBase + ni * 16 + lrow;
      const _Float16* Bp = Bw + (size_t)n * K + k0 + lhalf * 8;
      const half8 bl8 = *(const half8*)Bp;
      const half8 bh8 = *(const half8*)(Bp + 16);
      v16h bfrag = __builtin_shufflevector(bl8, bh8, 0, 1, 2, 3, 4, 5, 6, 7, 8,
                                           9, 10, 11, 12, 13, 14, 15);
#pragma unroll
      for (int mi = 0; mi < 2; ++mi)
        acc[mi][ni] = __builtin_amdgcn_wmma_f32_16x16x32_f16(
            false, afrag[mi], false, bfrag, (short)0, acc[mi][ni], false,
            false);
    }
  }

  // Epilogue: C/D layout — VGPR r, lane l -> M = r + (l>>4)*8, N = l&15.
  // One base address per (mi,ni); r-dimension becomes immediate offsets.
#pragma unroll
  for (int mi = 0; mi < 2; ++mi) {
    const int row0 = rowBase + mi * 16 + lhalf * 8;
#pragma unroll
    for (int ni = 0; ni < 4; ++ni) {
      const int col = colBase + ni * 16 + lrow;
      float bv = 0.0f;
      if (HASB) bv = bias[col];
      const size_t base = (size_t)row0 * N + col;
#pragma unroll
      for (int r = 0; r < 8; ++r) {
        const float v = acc[mi][ni][r] + bv;
        if (OUTF) outF[base + (size_t)r * N] = v;
        if (OUTH) outH[base + (size_t)r * N] = (_Float16)v;
      }
    }
  }
}

// ---------------------------------------------------------------------------
// Elementwise kernels
// ---------------------------------------------------------------------------
__device__ __forceinline__ float sigf(float x) { return 1.0f / (1.0f + __expf(-x)); }

// cell1 + boundary gate. g1: [B, 4224] = i|f|g|o|zpre.
__global__ __launch_bounds__(256) void e1_kernel(
    const float* __restrict__ g1, const float* __restrict__ vs,
    float* __restrict__ c1, float* __restrict__ h1f,
    _Float16* __restrict__ acat1, _Float16* __restrict__ acat2) {
  const int b = blockIdx.x;
  const float* g = g1 + (size_t)b * N1_;
  __shared__ float red[256];
  __shared__ float s_sh;
  float p = 0.0f;
  if (threadIdx.x < MID_) p = g[4 * HID_ + threadIdx.x] * vs[threadIdx.x];
  red[threadIdx.x] = p;
  __syncthreads();
  for (int s = 128; s > 0; s >>= 1) {
    if (threadIdx.x < s) red[threadIdx.x] += red[threadIdx.x + s];
    __syncthreads();
  }
  if (threadIdx.x == 0) s_sh = (red[0] > 0.0f) ? 1.0f : 0.0f;  // sigmoid(z)>0.5
  __syncthreads();
  const float s = s_sh;
  for (int n = threadIdx.x; n < HID_; n += 256) {
    const float ig = sigf(g[n]);
    const float fg = sigf(g[HID_ + n]);
    const float gg = tanhf(g[2 * HID_ + n]);
    const float og = sigf(g[3 * HID_ + n]);
    const float cn = fg * c1[(size_t)b * HID_ + n] + ig * gg;
    const float hn = og * tanhf(cn);
    acat2[(size_t)b * K2_ + n] = (_Float16)(hn * s);          // x2 = h1n*s
    const float h1 = hn * (1.0f - s);
    c1[(size_t)b * HID_ + n]  = cn * (1.0f - s);
    h1f[(size_t)b * HID_ + n] = h1;                            // output state
    acat1[(size_t)b * K1_ + PROJ_ + n] = (_Float16)h1;         // next-step A
  }
}

// cell2. g2: [B, 4096] = i|f|g|o.
__global__ __launch_bounds__(256) void e2_kernel(
    const float* __restrict__ g2, float* __restrict__ c2,
    _Float16* __restrict__ acat2) {
  const int b = blockIdx.x;
  const float* g = g2 + (size_t)b * N2_;
  for (int n = threadIdx.x; n < HID_; n += 256) {
    const float ig = sigf(g[n]);
    const float fg = sigf(g[HID_ + n]);
    const float gg = tanhf(g[2 * HID_ + n]);
    const float og = sigf(g[3 * HID_ + n]);
    const float cn = fg * c2[(size_t)b * HID_ + n] + ig * gg;
    c2[(size_t)b * HID_ + n] = cn;
    acat2[(size_t)b * K2_ + HID_ + n] = (_Float16)(og * tanhf(cn));  // h2
  }
}

// gather v_t (f16) into Acat1 cols [0,512): vectorized b128 copies.
__global__ __launch_bounds__(256) void gather_vt(const half8* __restrict__ v16,
                                                 half8* __restrict__ acat1,
                                                 int t) {
  const int idx = blockIdx.x * 256 + threadIdx.x;  // 0..B*PROJ/8-1
  const int b = idx >> 6, k = idx & 63;            // 64 half8 per row
  acat1[(size_t)b * (K1_ / 8) + k] = v16[((size_t)b * TT_ + t) * (PROJ_ / 8) + k];
}

// ---------------------------------------------------------------------------
// Packing / init kernels (run once per call)
// ---------------------------------------------------------------------------
__global__ __launch_bounds__(256) void cvt_f16(const float* __restrict__ s,
                                               _Float16* __restrict__ d, int n) {
  const int i = blockIdx.x * 256 + threadIdx.x;
  if (i < n) d[i] = (_Float16)s[i];
}

__global__ __launch_bounds__(256) void pack_wcat1(
    const float* __restrict__ W1_ih, const float* __restrict__ W1_hh,
    const float* __restrict__ Wsi, const float* __restrict__ Wsh,
    _Float16* __restrict__ dst) {
  const int i = blockIdx.x * 256 + threadIdx.x;
  if (i >= N1_ * K1_) return;
  const int r = i / K1_, c = i % K1_;
  float v;
  if (r < 4 * HID_)
    v = (c < PROJ_) ? W1_ih[(size_t)r * PROJ_ + c]
                    : W1_hh[(size_t)r * HID_ + (c - PROJ_)];
  else {
    const int rr = r - 4 * HID_;
    v = (c < PROJ_) ? Wsi[(size_t)rr * PROJ_ + c]
                    : Wsh[(size_t)rr * HID_ + (c - PROJ_)];
  }
  dst[i] = (_Float16)v;
}

__global__ __launch_bounds__(256) void pack_wcat2(
    const float* __restrict__ W2_ih, const float* __restrict__ W2_hh,
    _Float16* __restrict__ dst) {
  const int i = blockIdx.x * 256 + threadIdx.x;
  if (i >= N2_ * K2_) return;
  const int r = i / K2_, c = i % K2_;
  const float v = (c < HID_) ? W2_ih[(size_t)r * HID_ + c]
                             : W2_hh[(size_t)r * HID_ + (c - HID_)];
  dst[i] = (_Float16)v;
}

__global__ __launch_bounds__(256) void pack_bcat(const float* __restrict__ b1i,
                                                 const float* __restrict__ b1h,
                                                 const float* __restrict__ bbd,
                                                 float* __restrict__ dst) {
  const int i = blockIdx.x * 256 + threadIdx.x;
  if (i >= N1_) return;
  dst[i] = (i < 4 * HID_) ? (b1i[i] + b1h[i]) : bbd[i - 4 * HID_];
}

__global__ __launch_bounds__(256) void zero_u128(uint4* __restrict__ p, int n16) {
  const int i = blockIdx.x * 256 + threadIdx.x;
  if (i < n16) p[i] = uint4{0u, 0u, 0u, 0u};
}

// ---------------------------------------------------------------------------
extern "C" void kernel_launch(void* const* d_in, const int* in_sizes, int n_in,
                              void* d_out, int out_size, void* d_ws,
                              size_t ws_size, hipStream_t stream) {
  const float* video = (const float*)d_in[0];   // [B,T,FRAME]
  const float* We    = (const float*)d_in[1];   // [PROJ,FRAME]
  const float* be    = (const float*)d_in[2];   // [PROJ]
  const float* W1_ih = (const float*)d_in[3];   // [4H,PROJ]
  const float* W1_hh = (const float*)d_in[4];   // [4H,HID]
  const float* b1_ih = (const float*)d_in[5];
  const float* b1_hh = (const float*)d_in[6];
  const float* W2_ih = (const float*)d_in[7];   // [4H,HID]
  const float* W2_hh = (const float*)d_in[8];   // [4H,HID]
  const float* Wsi   = (const float*)d_in[9];   // [MID,PROJ]
  const float* Wsh   = (const float*)d_in[10];  // [MID,HID]
  const float* bbd   = (const float*)d_in[11];  // [MID]
  const float* vs    = (const float*)d_in[12];  // [1,MID]

  // ---- workspace carve-up -------------------------------------------------
  char* ws = (char*)d_ws;
  size_t off = 0;
  auto carve = [&](size_t bytes) {
    void* p = ws + off;
    off = (off + bytes + 255) & ~(size_t)255;
    return p;
  };
  _Float16* vH    = (_Float16*)carve((size_t)BB_ * TT_ * PROJ_ * 2);  // 25 MB
  _Float16* WeH   = (_Float16*)carve((size_t)PROJ_ * FRAME_ * 2);
  _Float16* Wc1H  = (_Float16*)carve((size_t)N1_ * K1_ * 2);          // 13 MB
  _Float16* Wc2H  = (_Float16*)carve((size_t)N2_ * K2_ * 2);          // 17 MB
  float*    bcat  = (float*)carve((size_t)N1_ * 4);
  _Float16* acat1 = (_Float16*)carve((size_t)BB_ * K1_ * 2);
  _Float16* acat2 = (_Float16*)carve((size_t)BB_ * K2_ * 2);
  float*    g1    = (float*)carve((size_t)BB_ * N1_ * 4);             // 8.7 MB
  float*    g2    = (float*)carve((size_t)BB_ * N2_ * 4);             // 8.4 MB
  float*    c1    = (float*)carve((size_t)BB_ * HID_ * 4);
  float*    c2    = (float*)carve((size_t)BB_ * HID_ * 4);
  float*    h1f   = (float*)carve((size_t)BB_ * HID_ * 4);

  auto blks = [](int n) { return (n + 255) / 256; };

  // ---- one-time (per call) prep ------------------------------------------
  cvt_f16<<<blks(PROJ_ * FRAME_), 256, 0, stream>>>(We, WeH, PROJ_ * FRAME_);
  pack_wcat1<<<blks(N1_ * K1_), 256, 0, stream>>>(W1_ih, W1_hh, Wsi, Wsh, Wc1H);
  pack_wcat2<<<blks(N2_ * K2_), 256, 0, stream>>>(W2_ih, W2_hh, Wc2H);
  pack_bcat<<<blks(N1_), 256, 0, stream>>>(b1_ih, b1_hh, bbd, bcat);
  // zero state: c1,c2,h1f (f32) and acat1,acat2 (f16), all 16B-multiples.
  zero_u128<<<blks(BB_ * HID_ / 4), 256, 0, stream>>>((uint4*)c1, BB_ * HID_ / 4);
  zero_u128<<<blks(BB_ * HID_ / 4), 256, 0, stream>>>((uint4*)c2, BB_ * HID_ / 4);
  zero_u128<<<blks(BB_ * HID_ / 4), 256, 0, stream>>>((uint4*)h1f, BB_ * HID_ / 4);
  zero_u128<<<blks(BB_ * K1_ / 8), 256, 0, stream>>>((uint4*)acat1, BB_ * K1_ / 8);
  zero_u128<<<blks(BB_ * K2_ / 8), 256, 0, stream>>>((uint4*)acat2, BB_ * K2_ / 8);

  // Frame embedding: v = video @ We^T + be  (f32 A, f16 out), M = B*T.
  {
    dim3 grid(PROJ_ / 128, (BB_ * TT_) / 128);
    wmma_gemm<true, true, false, true><<<grid, 256, 0, stream>>>(
        video, WeH, be, (float*)nullptr, vH, PROJ_, FRAME_, FRAME_);
  }

  // ---- sequential scan ----------------------------------------------------
  const dim3 grid1(N1_ / 128, BB_ / 128);
  const dim3 grid2(N2_ / 128, BB_ / 128);
  for (int t = 0; t < TT_; ++t) {
    gather_vt<<<blks(BB_ * PROJ_ / 8), 256, 0, stream>>>((half8*)vH,
                                                         (half8*)acat1, t);
    wmma_gemm<false, true, true, false><<<grid1, 256, 0, stream>>>(
        acat1, Wc1H, bcat, g1, (_Float16*)nullptr, N1_, K1_, K1_);
    e1_kernel<<<BB_, 256, 0, stream>>>(g1, vs, c1, h1f, acat1, acat2);
    wmma_gemm<false, false, true, false><<<grid2, 256, 0, stream>>>(
        acat2, Wc2H, (const float*)nullptr, g2, (_Float16*)nullptr, N2_, K2_,
        K2_);
    e2_kernel<<<BB_, 256, 0, stream>>>(g2, c2, acat2);
  }

  hipMemcpyAsync(d_out, h1f, (size_t)BB_ * HID_ * sizeof(float),
                 hipMemcpyDeviceToDevice, stream);
}
